// Attention_45114336478019
// MI455X (gfx1250) — matrix operand order repaired and checked by reading, not compiled, not run on hardware
//
#include <hip/hip_runtime.h>
#include <hip/hip_bf16.h>
#include <math.h>

// Problem constants (match reference)
#define B_    2
#define S_    1024
#define HID_  2048
#define NH_   16
#define NKV_  4
#define HD_   128   // head dim
#define NREP_ 4     // NH_/NKV_

typedef __attribute__((ext_vector_type(16))) __bf16       v16bf;
typedef __attribute__((ext_vector_type(8)))  float        v8f;
typedef __attribute__((ext_vector_type(4)))  unsigned int u32x4;

union FragU { u32x4 u[2]; v16bf v; };   // 32 bytes = one WMMA A/B bf16 fragment per lane

// float -> bf16, round-to-nearest-even (pure integer ops; always compiles)
__device__ inline __bf16 f2bf(float f) {
    unsigned u = __float_as_uint(f);
    unsigned r = (u + 0x7FFFu + ((u >> 16) & 1u)) >> 16;
    unsigned short h = (unsigned short)r;
    __bf16 b;
    __builtin_memcpy(&b, &h, 2);
    return b;
}

__device__ inline v8f wmma_bf16(const FragU& a, const FragU& b, v8f c) {
    return __builtin_amdgcn_wmma_f32_16x16x32_bf16(
        false, a.v, false, b.v, (short)0, c, false, false);
}

// ---------------------------------------------------------------------------
// Elementwise f32 -> bf16 conversion
// ---------------------------------------------------------------------------
__global__ void cvt_f32_bf16_kernel(const float* __restrict__ in,
                                    __bf16* __restrict__ out, int n) {
    int i = blockIdx.x * blockDim.x + threadIdx.x;
    int stride = gridDim.x * blockDim.x;
    for (; i < n; i += stride) out[i] = f2bf(in[i]);
}

// ---------------------------------------------------------------------------
// bf16 GEMM: C[M,N] (f32) = A[M,K] (bf16, row-major) @ W[K,N] (bf16, row-major)
// 8 waves (256 thr) per block; each wave computes a 32x32 tile:
//   2 A-frags + 2 B-frags -> 4 WMMAs per 32-wide K step (2x the WMMA/byte of a
//   16x16 tile). Block covers 64 rows x 128 cols.
// Fragment layouts per CDNA5 ISA 7.12.2 (16-bit A 16x32, B 32x16).
// ---------------------------------------------------------------------------
__global__ void gemm_bf16_kernel(const __bf16* __restrict__ A,
                                 const __bf16* __restrict__ W,
                                 float* __restrict__ C, int M, int N, int K) {
    const int wave = threadIdx.x >> 5;
    const int lane = threadIdx.x & 31;
    const int half = lane >> 4;   // 0: lanes 0-15, 1: lanes 16-31
    const int l16  = lane & 15;
    const int wr   = wave >> 2;   // 0..1  (row group)
    const int wc   = wave & 3;    // 0..3  (col group)
    const int m0 = blockIdx.x * 64 + wr * 32;
    const int n0 = blockIdx.y * 128 + wc * 32;
    if (m0 >= M || n0 >= N) return;           // wave-uniform guard (EXEC stays full)

    v8f acc00 = {}, acc01 = {}, acc10 = {}, acc11 = {};
    // A-frag: lane row = m0+l16 (+16); elements 0-7 -> k0+{0..7|8..15}, 8-15 -> +16
    const __bf16* ar0 = A + (size_t)(m0 + l16) * K + (half ? 8 : 0);
    const __bf16* ar1 = ar0 + (size_t)16 * K;
    // B-frag: lane column = n0+l16 (+16); elements over k in A's (table) order
    const __bf16* wrow = W + (size_t)n0 + l16;

    for (int k0 = 0; k0 < K; k0 += 32) {
        FragU a0, a1, b0, b1;
        a0.u[0] = *(const u32x4*)(ar0 + k0);
        a0.u[1] = *(const u32x4*)(ar0 + k0 + 16);
        a1.u[0] = *(const u32x4*)(ar1 + k0);
        a1.u[1] = *(const u32x4*)(ar1 + k0 + 16);
        const __bf16* wp = wrow + (size_t)k0 * N;
#pragma unroll
        for (int s = 0; s < 16; ++s) {
            int kk = (s < 8) ? (8 * half + s) : (16 + 8 * half + s - 8);
            b0.v[s] = wp[(size_t)kk * N];
            b1.v[s] = wp[(size_t)kk * N + 16];
        }
        // cover latency one K-step ahead (global_prefetch_b8)
        __builtin_prefetch(ar0 + k0 + 32, 0, 0);
        __builtin_prefetch(ar1 + k0 + 32, 0, 0);
        __builtin_prefetch(wp + (size_t)32 * N, 0, 0);
        acc00 = wmma_bf16(a0, b0, acc00);
        acc01 = wmma_bf16(a0, b1, acc01);
        acc10 = wmma_bf16(a1, b0, acc10);
        acc11 = wmma_bf16(a1, b1, acc11);
    }
    // C layout: lane element j -> row +j+8*half, col +l16
    float* c0 = C + (size_t)(m0 + 8 * half) * N + n0 + l16;
    float* c1 = C + (size_t)(m0 + 16 + 8 * half) * N + n0 + l16;
#pragma unroll
    for (int j = 0; j < 8; ++j) {
        c0[(size_t)j * N]      = acc00[j];
        c0[(size_t)j * N + 16] = acc01[j];
        c1[(size_t)j * N]      = acc10[j];
        c1[(size_t)j * N + 16] = acc11[j];
    }
}

// ---------------------------------------------------------------------------
// RoPE on Q: Qf [B*S, HID] f32 -> Qbf [B, NH, S, HD] bf16
// ---------------------------------------------------------------------------
__global__ void rope_q_kernel(const float* __restrict__ Qf,
                              const float* __restrict__ cosb,
                              const float* __restrict__ sinb,
                              __bf16* __restrict__ Qbf) {
    int idx = blockIdx.x * blockDim.x + threadIdx.x;   // B*S*NH*64 = 2^21 threads
    int hd = idx & 63;
    int h  = (idx >> 6) & (NH_ - 1);
    int s  = (idx >> 10) & (S_ - 1);
    int b  = idx >> 20;
    if (b >= B_) return;
    const float* qrow = Qf + (size_t)(b * S_ + s) * HID_ + h * HD_;
    float q1 = qrow[hd], q2 = qrow[hd + 64];
    const float* cr = cosb + (size_t)s * HD_;
    const float* sr = sinb + (size_t)s * HD_;
    float o1 = q1 * cr[hd]      - q2 * sr[hd];
    float o2 = q2 * cr[hd + 64] + q1 * sr[hd + 64];
    __bf16* orow = Qbf + ((size_t)(b * NH_ + h) * S_ + s) * HD_;
    orow[hd]      = f2bf(o1);
    orow[hd + 64] = f2bf(o2);
}

// ---------------------------------------------------------------------------
// RoPE on K, writing K TRANSPOSED: Kf [B*S, NKV*HD] f32 -> Kt [B, NKV, HD, S] bf16
// (d-major so score-WMMA B-fragments are contiguous along keys)
// ---------------------------------------------------------------------------
__global__ void rope_k_kernel(const float* __restrict__ Kf,
                              const float* __restrict__ cosb,
                              const float* __restrict__ sinb,
                              __bf16* __restrict__ Kt) {
    int idx = blockIdx.x * blockDim.x + threadIdx.x;   // B*S*NKV*64 = 2^19 threads
    int hd = idx & 63;
    int kv = (idx >> 6) & (NKV_ - 1);
    int s  = (idx >> 8) & (S_ - 1);
    int b  = idx >> 18;
    if (b >= B_) return;
    const float* krow = Kf + (size_t)(b * S_ + s) * (NKV_ * HD_) + kv * HD_;
    float k1 = krow[hd], k2 = krow[hd + 64];
    const float* cr = cosb + (size_t)s * HD_;
    const float* sr = sinb + (size_t)s * HD_;
    float o1 = k1 * cr[hd]      - k2 * sr[hd];
    float o2 = k2 * cr[hd + 64] + k1 * sr[hd + 64];
    __bf16* ob = Kt + (size_t)(b * NKV_ + kv) * HD_ * S_;
    ob[(size_t)hd * S_ + s]        = f2bf(o1);
    ob[(size_t)(hd + 64) * S_ + s] = f2bf(o2);
}

// ---------------------------------------------------------------------------
// V relayout: Vf [B*S, NKV*HD] f32 -> Vbf [B, NKV, S, HD] bf16
// ---------------------------------------------------------------------------
__global__ void cvt_v_kernel(const float* __restrict__ Vf,
                             __bf16* __restrict__ Vbf) {
    int idx = blockIdx.x * blockDim.x + threadIdx.x;   // B*S*NKV*HD = 2^20 threads
    int d  = idx & (HD_ - 1);
    int kv = (idx >> 7) & (NKV_ - 1);
    int s  = (idx >> 9) & (S_ - 1);
    int b  = idx >> 19;
    if (b >= B_) return;
    Vbf[((size_t)(b * NKV_ + kv) * S_ + s) * HD_ + d] =
        f2bf(Vf[(size_t)(b * S_ + s) * (NKV_ * HD_) + kv * HD_ + d]);
}

// ---------------------------------------------------------------------------
// Flash attention (causal, GQA): 4 waves per 128-thread block, each wave owns
// one (b, h, 16-query tile) and a private 1 KB LDS slice. No block barriers:
// the P-tile LDS round-trip is intra-wave only (per-wave in-order LDS +
// explicit s_wait_dscnt), so divergent causal trip counts between waves are
// safe. Scores via 8 bf16 WMMAs / 32-key block, online softmax via half-wave
// shuffles, P.V via 8 bf16 WMMAs into fp32 accumulators.
// AO written as [B*S, HID] f32 (row = b*S+q, col = h*HD+d) for the O-GEMM.
// ---------------------------------------------------------------------------
__global__ void attn_kernel(const __bf16* __restrict__ Q,
                            const __bf16* __restrict__ Kt,
                            const __bf16* __restrict__ V,
                            float* __restrict__ AO) {
    __shared__ __bf16 Plds[4][16 * 32];     // one 16x32 P tile per wave

    const int wave = threadIdx.x >> 5;      // 0..3
    const int lane = threadIdx.x & 31;
    const int half = lane >> 4;
    const int l16  = lane & 15;
    const int qt = blockIdx.x * 4 + wave;
    const int h = blockIdx.y, b = blockIdx.z;
    const int kv = h >> 2;            // NREP_ = 4
    const int q0 = qt * 16;
    __bf16* myP = Plds[wave];

    // Q A-fragments, 4 chunks of K=32 covering HD=128
    FragU qa[4];
    const __bf16* qrow =
        Q + ((size_t)(b * NH_ + h) * S_ + q0 + l16) * HD_ + (half ? 8 : 0);
#pragma unroll
    for (int c = 0; c < 4; ++c) {
        qa[c].u[0] = *(const u32x4*)(qrow + c * 32);
        qa[c].u[1] = *(const u32x4*)(qrow + c * 32 + 16);
    }

    const __bf16* ktb = Kt + (size_t)(b * NKV_ + kv) * HD_ * S_;
    const __bf16* vb  = V  + (size_t)(b * NKV_ + kv) * S_ * HD_;

    v8f out[8];
    v8f vzero = {};
#pragma unroll
    for (int dc = 0; dc < 8; ++dc) out[dc] = vzero;
    float mrow[8], lrow[8];
#pragma unroll
    for (int j = 0; j < 8; ++j) { mrow[j] = -__builtin_inff(); lrow[j] = 0.f; }

    const float scale = 0.08838834764831845f;   // 1/sqrt(128)
    const int dlane = l16 + 16 * half;          // lane's K index within 32-chunk
    const int nkb = (q0 + 16 + 31) / 32;        // causal bound, 32-key blocks

    for (int kb = 0; kb < nkb; ++kb) {
        const int key0 = kb * 32;
        v8f sa = vzero, sb = vzero;             // two 16-key score tiles
#pragma unroll
        for (int c = 0; c < 4; ++c) {
            const __bf16* kp = ktb + (size_t)(c * 32) * S_ + key0 + l16;
            FragU bk, bk2;
#pragma unroll
            for (int s = 0; s < 16; ++s) {
                int kk = (s < 8) ? (8 * half + s) : (16 + 8 * half + s - 8);
                bk.v[s]  = kp[(size_t)kk * S_];
                bk2.v[s] = kp[(size_t)kk * S_ + 16];
            }
            sa = wmma_bf16(qa[c], bk,  sa);
            sb = wmma_bf16(qa[c], bk2, sb);
        }

        // --- online softmax update (rows j+8*half, cols l16 / 16+l16) ---
        float fac[8], pa[8], pb[8];
#pragma unroll
        for (int j = 0; j < 8; ++j) {
            int qr = q0 + j + 8 * half;
            float va = sa[j] * scale;
            float vc = sb[j] * scale;
            if (key0 + l16 > qr)      va = -__builtin_inff();
            if (key0 + 16 + l16 > qr) vc = -__builtin_inff();
            float tm = fmaxf(va, vc);
#pragma unroll
            for (int mb = 1; mb <= 8; mb <<= 1) tm = fmaxf(tm, __shfl_xor(tm, mb, 32));
            float mnew = fmaxf(mrow[j], tm);
            fac[j] = __expf(mrow[j] - mnew);
            pa[j]  = __expf(va - mnew);
            pb[j]  = __expf(vc - mnew);
            float rs = pa[j] + pb[j];
#pragma unroll
            for (int mb = 1; mb <= 8; mb <<= 1) rs += __shfl_xor(rs, mb, 32);
            lrow[j] = lrow[j] * fac[j] + rs;
            mrow[j] = mnew;
        }
#pragma unroll
        for (int dc = 0; dc < 8; ++dc)
#pragma unroll
            for (int j = 0; j < 8; ++j) out[dc][j] *= fac[j];

        // --- P (16x32 bf16) through private LDS slice to A-fragment layout ---
        asm volatile("" ::: "memory");          // keep prior LDS reads before stores
#pragma unroll
        for (int j = 0; j < 8; ++j) {
            int r = j + 8 * half;
            myP[r * 32 + l16]      = f2bf(pa[j]);
            myP[r * 32 + 16 + l16] = f2bf(pb[j]);
        }
        // per-wave in-order LDS: wait stores, then re-stripe into A-fragment
        asm volatile("s_wait_dscnt 0x0" ::: "memory");
        FragU pf;
        pf.u[0] = *(const u32x4*)&myP[l16 * 32 + (half ? 8 : 0)];
        pf.u[1] = *(const u32x4*)&myP[l16 * 32 + (half ? 8 : 0) + 16];

        // --- P . V : B-frag lane column d = dc*16 + l16, elements over keys in A's k order ---
#pragma unroll
        for (int dc = 0; dc < 8; ++dc) {
            FragU bv;
#pragma unroll
            for (int s = 0; s < 16; ++s) {
                int key = key0 + ((s < 8) ? (8 * half + s) : (16 + 8 * half + s - 8));
                if (key >= S_) key = S_ - 1;    // OOB clamp; P==0 there anyway
                bv.v[s] = vb[(size_t)key * HD_ + dc * 16 + l16];
            }
            out[dc] = wmma_bf16(pf, bv, out[dc]);
        }
    }

    // finalize: divide by l, write [b*S+q, h*HD + d]
#pragma unroll
    for (int dc = 0; dc < 8; ++dc) {
#pragma unroll
        for (int j = 0; j < 8; ++j) {
            int qq = q0 + j + 8 * half;
            AO[(size_t)(b * S_ + qq) * HID_ + h * HD_ + dc * 16 + l16] =
                out[dc][j] / lrow[j];
        }
    }
}

// ---------------------------------------------------------------------------
// Host orchestration
// ---------------------------------------------------------------------------
extern "C" void kernel_launch(void* const* d_in, const int* in_sizes, int n_in,
                              void* d_out, int out_size, void* d_ws, size_t ws_size,
                              hipStream_t stream) {
    const float* x    = (const float*)d_in[0];
    const float* cosb = (const float*)d_in[1];
    const float* sinb = (const float*)d_in[2];
    const float* wq   = (const float*)d_in[3];
    const float* wk   = (const float*)d_in[4];
    const float* wv   = (const float*)d_in[5];
    const float* wo   = (const float*)d_in[6];
    float* out = (float*)d_out;

    const size_t MS  = (size_t)B_ * S_;        // 2048 rows
    const size_t KVW = (size_t)HID_ * NKV_ * HD_;

    char* p = (char*)d_ws;
    auto alloc = [&](size_t bytes) -> char* {
        char* r = p;
        p += (bytes + 255) & ~(size_t)255;
        return r;
    };
    __bf16* xbf  = (__bf16*)alloc(MS * HID_ * 2);
    __bf16* wqbf = (__bf16*)alloc((size_t)HID_ * HID_ * 2);
    __bf16* wkbf = (__bf16*)alloc(KVW * 2);
    __bf16* wvbf = (__bf16*)alloc(KVW * 2);
    __bf16* wobf = (__bf16*)alloc((size_t)HID_ * HID_ * 2);
    float*  Qf   = (float*)alloc(MS * HID_ * 4);
    float*  Kf   = (float*)alloc(MS * NKV_ * HD_ * 4);
    float*  Vf   = (float*)alloc(MS * NKV_ * HD_ * 4);
    __bf16* Qbf  = (__bf16*)alloc(MS * HID_ * 2);
    __bf16* Ktbf = (__bf16*)alloc(MS * NKV_ * HD_ * 2);
    __bf16* Vbf  = (__bf16*)alloc(MS * NKV_ * HD_ * 2);
    float*  AOf  = (float*)alloc(MS * HID_ * 4);
    __bf16* AObf = (__bf16*)alloc(MS * HID_ * 2);

    dim3 blk(256);
    auto cvt = [&](const float* i, __bf16* o, size_t n) {
        cvt_f32_bf16_kernel<<<dim3((unsigned)((n + 255) / 256)), blk, 0, stream>>>(i, o, (int)n);
    };

    // 1) bf16 copies of activations + weights
    cvt(x,  xbf,  MS * HID_);
    cvt(wq, wqbf, (size_t)HID_ * HID_);
    cvt(wk, wkbf, KVW);
    cvt(wv, wvbf, KVW);
    cvt(wo, wobf, (size_t)HID_ * HID_);

    // 2) Q/K/V projections (bf16 WMMA GEMMs, f32 out); blocks cover 64x128
    gemm_bf16_kernel<<<dim3(MS / 64, HID_ / 128), blk, 0, stream>>>(
        xbf, wqbf, Qf, (int)MS, HID_, HID_);
    gemm_bf16_kernel<<<dim3(MS / 64, (NKV_ * HD_) / 128), blk, 0, stream>>>(
        xbf, wkbf, Kf, (int)MS, NKV_ * HD_, HID_);
    gemm_bf16_kernel<<<dim3(MS / 64, (NKV_ * HD_) / 128), blk, 0, stream>>>(
        xbf, wvbf, Vf, (int)MS, NKV_ * HD_, HID_);

    // 3) RoPE + relayout to attention-friendly bf16 layouts
    rope_q_kernel<<<dim3((B_ * S_ * NH_ * 64) / 256), blk, 0, stream>>>(Qf, cosb, sinb, Qbf);
    rope_k_kernel<<<dim3((B_ * S_ * NKV_ * 64) / 256), blk, 0, stream>>>(Kf, cosb, sinb, Ktbf);
    cvt_v_kernel<<<dim3((B_ * S_ * NKV_ * HD_) / 256), blk, 0, stream>>>(Vf, Vbf);

    // 4) causal flash attention (4 q-tiles per 128-thread block)
    attn_kernel<<<dim3(S_ / 64, NH_, B_), dim3(128), 0, stream>>>(Qbf, Ktbf, Vbf, AOf);

    // 5) output projection
    cvt(AOf, AObf, MS * HID_);
    gemm_bf16_kernel<<<dim3(MS / 64, HID_ / 128), blk, 0, stream>>>(
        AObf, wobf, out, (int)MS, HID_, HID_);
}
